// LinearAttention_19945828123012
// MI455X (gfx1250) — matrix operand order repaired and checked by reading, not compiled, not run on hardware
//
#include <hip/hip_runtime.h>
#include <hip/hip_bf16.h>

// ---------------------------------------------------------------------------
// Causal linear attention for MI455X (gfx1250). All matmuls are
// v_wmma_f32_16x16x32_bf16 (wave32). All conversions/transposes are hoisted
// out of the GEMM k-loops: operand staging is pure float4 global->LDS copy,
// double-buffered so next-step loads fly under the current WMMAs.
//
//   0) convert_x / convert_wT : one-shot f32->bf16 (+transpose for weights)
//   1) qkv_gemm_kernel : xb @ WqkvT + b -> q[t][e], k[t][e], kT[e][t], vT[e][t]
//   2) attn_kernel     : per (b,h) chunked (C=64) causal linear-attn scan
//   3) out_gemm_kernel : O @ WoutT + b -> f32 d_out
// ---------------------------------------------------------------------------

typedef __attribute__((ext_vector_type(16))) __bf16 v16bf;
typedef __attribute__((ext_vector_type(8)))  float  v8f;

__device__ __forceinline__ v8f vzero8() {
  v8f z;
#pragma unroll
  for (int i = 0; i < 8; ++i) z[i] = 0.0f;
  return z;
}

__device__ __forceinline__ v8f wmma_bf16(v16bf a, v16bf b, v8f c) {
  return __builtin_amdgcn_wmma_f32_16x16x32_bf16(
      false, a, false, b, (short)0, c, false, false);
}

// A fragment: 16x32 bf16 row-major tile; two contiguous 16B runs per lane.
__device__ __forceinline__ v16bf load_frag_a(const __bf16* base, int ld) {
  const int lane = threadIdx.x & 31;
  const __bf16* p = base + (lane & 15) * ld + (lane >> 4) * 8;
  v16bf a;
  ((float4*)&a)[0] = *(const float4*)(p);
  ((float4*)&a)[1] = *(const float4*)(p + 16);
  return a;
}

// B fragment from transposed staging: B[k][n] = base[n*ld + k]; two 16B runs/lane.
__device__ __forceinline__ v16bf load_frag_bT(const __bf16* base, int ld) {
  const int lane = threadIdx.x & 31;
  const __bf16* p = base + (lane & 15) * ld + (lane >> 4) * 8;
  v16bf b;
  ((float4*)&b)[0] = *(const float4*)(p);
  ((float4*)&b)[1] = *(const float4*)(p + 16);
  return b;
}

__device__ __forceinline__ float elu1(float x) {
  return (x > 0.0f ? x : (__expf(x) - 1.0f)) + 1.0001f;   // elu(x)+1+1e-4
}

// ---------------------------------------------------------------------------
// One-shot converters (streaming; cost ~4us of the 23.3 TB/s budget)
// ---------------------------------------------------------------------------
__global__ __launch_bounds__(256) void convert_x_kernel(
    const float* __restrict__ X, __bf16* __restrict__ Xb) {
  const size_t base = ((size_t)blockIdx.x * 256 + threadIdx.x) * 32;
#pragma unroll
  for (int it = 0; it < 4; ++it) {
    float4 a = *(const float4*)(X + base + it * 8);
    float4 b = *(const float4*)(X + base + it * 8 + 4);
    __bf16 t[8];
    t[0] = (__bf16)a.x; t[1] = (__bf16)a.y; t[2] = (__bf16)a.z; t[3] = (__bf16)a.w;
    t[4] = (__bf16)b.x; t[5] = (__bf16)b.y; t[6] = (__bf16)b.z; t[7] = (__bf16)b.w;
    *(float4*)(Xb + base + it * 8) = *(const float4*)t;
  }
}

// W (f32, KxN row-major) -> WT (bf16, NxK row-major), 32x32 LDS tile transpose
__global__ __launch_bounds__(256) void convert_wT_kernel(
    const float* __restrict__ W, __bf16* __restrict__ WT, int K, int N) {
  __shared__ __align__(16) __bf16 T[32 * 48];
  const int tid = threadIdx.x;
  const int n0 = blockIdx.x * 32, k0 = blockIdx.y * 32;
#pragma unroll
  for (int it = 0; it < 4; ++it) {
    const int f = tid + it * 256;           // 0..1023
    const int kr = f >> 5, nc = f & 31;
    T[nc * 48 + kr] = (__bf16)W[(size_t)(k0 + kr) * N + n0 + nc];
  }
  __syncthreads();
  if (tid < 128) {
    const int n = tid >> 2, k8 = tid & 3;
    *(float4*)(WT + (size_t)(n0 + n) * K + k0 + k8 * 8) =
        *(const float4*)(T + n * 48 + k8 * 8);
  }
}

// ---------------------------------------------------------------------------
// Kernel 1: qkv = xb @ WqkvT + b ; feature map; outputs q,k row-major
// [bh][t][e] plus kT,vT transposed [bh][e][t]. M=16384,N=3072,K=1024.
// 256x64 block tile; wave owns 32x64 (2x4 tiles); double-buffered LDS.
// ---------------------------------------------------------------------------
__global__ __launch_bounds__(256) void qkv_gemm_kernel(
    const __bf16* __restrict__ A, const __bf16* __restrict__ WT,
    const float* __restrict__ bias,
    __bf16* __restrict__ Qo, __bf16* __restrict__ Ko,
    __bf16* __restrict__ KTo, __bf16* __restrict__ VTo) {
  constexpr int ALD = 48, BLD = 48;
  constexpr int ASZ = 256 * ALD, BSZ = 64 * BLD;
  __shared__ __align__(16) __bf16 As[2 * ASZ];
  __shared__ __align__(16) __bf16 BsT[2 * BSZ];

  const int tid  = threadIdx.x;
  const int wave = tid >> 5, lane = tid & 31;
  const int col  = lane & 15, halfl = lane >> 4;
  const int nB = blockIdx.x;        // 0..47
  const int mB = blockIdx.y;        // 0..63

  v8f acc[2][4];
#pragma unroll
  for (int tt = 0; tt < 2; ++tt)
#pragma unroll
    for (int tc = 0; tc < 4; ++tc) acc[tt][tc] = vzero8();

  float4 av[4], wv;
  // prologue: k-step 0 -> buffer 0 (pure float4 copies, bf16 already)
#pragma unroll
  for (int it = 0; it < 4; ++it) {
    const int f = tid + it * 256, r = f >> 2, c8 = f & 3;
    av[it] = *(const float4*)(A + (size_t)(mB * 256 + r) * 1024 + c8 * 8);
  }
  { const int n = tid >> 2, c8 = tid & 3;
    wv = *(const float4*)(WT + (size_t)(nB * 64 + n) * 1024 + c8 * 8); }
#pragma unroll
  for (int it = 0; it < 4; ++it) {
    const int f = tid + it * 256, r = f >> 2, c8 = f & 3;
    *(float4*)(As + r * ALD + c8 * 8) = av[it];
  }
  { const int n = tid >> 2, c8 = tid & 3;
    *(float4*)(BsT + n * BLD + c8 * 8) = wv; }
  __syncthreads();

  for (int ks = 0; ks < 32; ++ks) {
    const int cur = ks & 1;
    if (ks + 1 < 32) {            // issue next k-step's loads (no conversion)
      const int kk = (ks + 1) * 32;
#pragma unroll
      for (int it = 0; it < 4; ++it) {
        const int f = tid + it * 256, r = f >> 2, c8 = f & 3;
        av[it] = *(const float4*)(A + (size_t)(mB * 256 + r) * 1024 + kk + c8 * 8);
      }
      const int n = tid >> 2, c8 = tid & 3;
      wv = *(const float4*)(WT + (size_t)(nB * 64 + n) * 1024 + kk + c8 * 8);
    }
    {
      v16bf a0 = load_frag_a(As + cur * ASZ + (wave * 2 + 0) * 16 * ALD, ALD);
      v16bf a1 = load_frag_a(As + cur * ASZ + (wave * 2 + 1) * 16 * ALD, ALD);
#pragma unroll
      for (int tc = 0; tc < 4; ++tc) {
        v16bf b = load_frag_bT(BsT + cur * BSZ + tc * 16 * BLD, BLD);
        acc[0][tc] = wmma_bf16(a0, b, acc[0][tc]);
        acc[1][tc] = wmma_bf16(a1, b, acc[1][tc]);
      }
    }
    if (ks + 1 < 32) {
      const int nxt = cur ^ 1;
#pragma unroll
      for (int it = 0; it < 4; ++it) {
        const int f = tid + it * 256, r = f >> 2, c8 = f & 3;
        *(float4*)(As + nxt * ASZ + r * ALD + c8 * 8) = av[it];
      }
      const int n = tid >> 2, c8 = tid & 3;
      *(float4*)(BsT + nxt * BSZ + n * BLD + c8 * 8) = wv;
    }
    __syncthreads();
  }

  // epilogue: bias + feature map; q,k scattered row-major, kT/vT as one
  // coalesced 16B store per lane per tile (transposed layout matches the
  // WMMA C layout: lane holds 8 consecutive rows of one column).
#pragma unroll
  for (int tt = 0; tt < 2; ++tt) {
    const int rgb = mB * 256 + (wave * 2 + tt) * 16 + halfl * 8;  // row base
    const int bb = rgb >> 12, l0 = rgb & 4095;
#pragma unroll
    for (int tc = 0; tc < 4; ++tc) {
      const int cg = nB * 64 + tc * 16 + col;        // 0..3071
      const float bv = bias[cg];
      const int which = cg >> 10;                    // 0=q 1=k 2=v
      const int d = cg & 1023, hh = d >> 6, e = d & 63;
      const size_t bh = (size_t)(bb * 16 + hh);
      if (which == 0) {
#pragma unroll
        for (int i = 0; i < 8; ++i)
          Qo[(bh * 4096 + l0 + i) * 64 + e] =
              (__bf16)elu1((acc[tt][tc][i] + bv) * 0.125f);   // E^-0.5 = 1/8
      } else if (which == 1) {
        __bf16 k8v[8];
#pragma unroll
        for (int i = 0; i < 8; ++i) {
          const float kf = elu1(acc[tt][tc][i] + bv);
          Ko[(bh * 4096 + l0 + i) * 64 + e] = (__bf16)kf;
          k8v[i] = (__bf16)kf;
        }
        *(float4*)(KTo + (bh * 64 + e) * 4096 + l0) = *(const float4*)k8v;
      } else {
        __bf16 v8v[8];
#pragma unroll
        for (int i = 0; i < 8; ++i) v8v[i] = (__bf16)(acc[tt][tc][i] + bv);
        *(float4*)(VTo + (bh * 64 + e) * 4096 + l0) = *(const float4*)v8v;
      }
    }
  }
}

// ---------------------------------------------------------------------------
// Kernel 2: chunked causal linear attention scan. One block per (b,h).
// ---------------------------------------------------------------------------
__global__ __launch_bounds__(256) void attn_kernel(
    const __bf16* __restrict__ Q, const __bf16* __restrict__ K,
    const __bf16* __restrict__ KT, const __bf16* __restrict__ VT,
    __bf16* __restrict__ O) {
  __shared__ __align__(16) __bf16 qs [64 * 64];   // [t][e]
  __shared__ __align__(16) __bf16 ks [64 * 64];   // [t][e]
  __shared__ __align__(16) __bf16 ksT[64 * 64];   // [e][t]
  __shared__ __align__(16) __bf16 vsT[64 * 64];   // [e][t]
  __shared__ __align__(16) __bf16 Sb [64 * 64];   // masked S [t][s]; reused as O
  __shared__ __align__(16) __bf16 KVbT[64 * 64];  // KV^T image [f][e]
  __shared__ float ksum[64];
  __shared__ float den[64];

  const int bh  = blockIdx.x;                      // b*16 + h
  const int b   = bh >> 4, h = bh & 15;
  const int tid = threadIdx.x;
  const int wave = tid >> 5, lane = tid & 31;
  const int col  = lane & 15, halfl = lane >> 4;

  v8f kvacc[2] = {vzero8(), vzero8()};             // persistent KV state tiles
  for (int i = tid; i < 4096; i += 256) KVbT[i] = (__bf16)0.0f;
  if (tid < 64) ksum[tid] = 0.0f;
  __syncthreads();

  const __bf16* Qb  = Q  + (size_t)bh * 4096 * 64;
  const __bf16* Kb  = K  + (size_t)bh * 4096 * 64;
  const __bf16* KTb = KT + (size_t)bh * 64 * 4096;
  const __bf16* VTb = VT + (size_t)bh * 64 * 4096;
  __bf16* Ob = O + (size_t)b * 4096 * 1024 + h * 64;

  for (int ch = 0; ch < 64; ++ch) {
    // ---- stage chunk: all pure float4 copies (transposes precomputed) ----
    float4 q4[2], k4[2], kt4[2], vt4[2];
#pragma unroll
    for (int it = 0; it < 2; ++it) {
      const int f = tid + it * 256;
      q4[it] = ((const float4*)(Qb + (size_t)ch * 4096))[f];
      k4[it] = ((const float4*)(Kb + (size_t)ch * 4096))[f];
      const int e = f >> 3, t8 = (f & 7) * 8;
      kt4[it] = *(const float4*)(KTb + (size_t)e * 4096 + ch * 64 + t8);
      vt4[it] = *(const float4*)(VTb + (size_t)e * 4096 + ch * 64 + t8);
    }
    if (ch + 1 < 64 && tid < 64) {  // prefetch next chunk (global_prefetch_b8)
      __builtin_prefetch(Qb  + (size_t)(ch + 1) * 4096 + tid * 64, 0, 1);
      __builtin_prefetch(Kb  + (size_t)(ch + 1) * 4096 + tid * 64, 0, 1);
      __builtin_prefetch(KTb + (size_t)tid * 4096 + (ch + 1) * 64, 0, 1);
      __builtin_prefetch(VTb + (size_t)tid * 4096 + (ch + 1) * 64, 0, 1);
    }
#pragma unroll
    for (int it = 0; it < 2; ++it) {
      const int f = tid + it * 256;
      ((float4*)qs)[f]  = q4[it];
      ((float4*)ks)[f]  = k4[it];
      ((float4*)ksT)[f] = kt4[it];
      ((float4*)vsT)[f] = vt4[it];
    }
    __syncthreads();

    // ---- den base: q_t . ksum_old (vectorized row read) ----
    if (tid < 64) {
      float dq = 0.0f;
      const __bf16* qr = qs + tid * 64;
#pragma unroll
      for (int e8 = 0; e8 < 8; ++e8) {
        float4 blk = *(const float4*)(qr + e8 * 8);
        const __bf16* bp = (const __bf16*)&blk;
#pragma unroll
        for (int j = 0; j < 8; ++j) dq += (float)bp[j] * ksum[e8 * 8 + j];
      }
      den[tid] = dq;
    }
    __syncthreads();

    // ---- S = q @ k^T, causal mask, row-sums into den ----
#pragma unroll
    for (int t = 0; t < 2; ++t) {
      const int tile = wave * 2 + t, tr = tile >> 2, tc = tile & 3;
      v8f acc = vzero8();
#pragma unroll
      for (int kb = 0; kb < 2; ++kb) {
        v16bf a  = load_frag_a (qs + tr * 16 * 64 + kb * 32, 64);
        v16bf bt = load_frag_bT(ks + tc * 16 * 64 + kb * 32, 64); // B[e][s]=k[s][e]
        acc = wmma_bf16(a, bt, acc);
      }
      const int n = tc * 16 + col;
#pragma unroll
      for (int i = 0; i < 8; ++i) {
        const int m = tr * 16 + halfl * 8 + i;
        const float sv = (n <= m) ? acc[i] : 0.0f;
        Sb[m * 64 + n] = (__bf16)sv;
        float r = sv;
        r += __shfl_xor(r, 1, 32);
        r += __shfl_xor(r, 2, 32);
        r += __shfl_xor(r, 4, 32);
        r += __shfl_xor(r, 8, 32);           // stays within 16-lane half
        if (col == 0) atomicAdd(&den[m], r); // ds_add_f32
      }
    }
    __syncthreads();

    // ---- O = S_masked @ v + q @ KV_old (regs) ; KV += k^T @ v ----
    v8f oacc[2];
#pragma unroll
    for (int t = 0; t < 2; ++t) {
      const int tile = wave * 2 + t, tr = tile >> 2, tc = tile & 3;
      v8f acc = vzero8();
#pragma unroll
      for (int kb = 0; kb < 2; ++kb) {
        v16bf a1 = load_frag_a (Sb + tr * 16 * 64 + kb * 32, 64);
        v16bf b1 = load_frag_bT(vsT + tc * 16 * 64 + kb * 32, 64);
        acc = wmma_bf16(a1, b1, acc);
        v16bf a2 = load_frag_a (qs + tr * 16 * 64 + kb * 32, 64);
        v16bf b2 = load_frag_bT(KVbT + tc * 16 * 64 + kb * 32, 64);
        acc = wmma_bf16(a2, b2, acc);
      }
      oacc[t] = acc;
#pragma unroll
      for (int kb = 0; kb < 2; ++kb) {
        v16bf a = load_frag_a (ksT + tr * 16 * 64 + kb * 32, 64);  // A[e][t]
        v16bf b = load_frag_bT(vsT + tc * 16 * 64 + kb * 32, 64);  // B[t][f]
        kvacc[t] = wmma_bf16(a, b, kvacc[t]);
      }
    }
    __syncthreads();   // all reads of Sb / KVbT complete

    // ---- write KVbT (16B/lane/tile), O into Sb, ksum update ----
#pragma unroll
    for (int t = 0; t < 2; ++t) {
      const int tile = wave * 2 + t, tr = tile >> 2, tc = tile & 3;
      const int n = tc * 16 + col;
      const int mb = tr * 16 + halfl * 8;
      __bf16 tmp[8];
#pragma unroll
      for (int i = 0; i < 8; ++i) tmp[i] = (__bf16)kvacc[t][i];
      *(float4*)(KVbT + n * 64 + mb) = *(const float4*)tmp;
#pragma unroll
      for (int i = 0; i < 8; ++i) {
        const int m = mb + i;
        Sb[m * 64 + n] = (__bf16)(oacc[t][i] / (den[m] + 1e-6f));
      }
    }
    if (tid < 64) {
      float s = ksum[tid];
      const __bf16* kr = ksT + tid * 64;
#pragma unroll
      for (int e8 = 0; e8 < 8; ++e8) {
        float4 blk = *(const float4*)(kr + e8 * 8);
        const __bf16* bp = (const __bf16*)&blk;
#pragma unroll
        for (int j = 0; j < 8; ++j) s += (float)bp[j];
      }
      ksum[tid] = s;
    }
    __syncthreads();

    // ---- coalesced 128-bit global stores of the O chunk ----
#pragma unroll
    for (int it = 0; it < 2; ++it) {
      const int f = tid + it * 256;        // 0..511
      const int r = f >> 3, c8 = f & 7;
      float4 d = *(const float4*)(Sb + r * 64 + c8 * 8);
      *(float4*)(Ob + (size_t)(ch * 64 + r) * 1024 + c8 * 8) = d;
    }
  }
}

// ---------------------------------------------------------------------------
// Kernel 3: out = O(bf16) @ WoutT(bf16) + b_out -> f32. M=16384,N=1024,K=1024.
// Same 256x64 double-buffered core as kernel 1.
// ---------------------------------------------------------------------------
__global__ __launch_bounds__(256) void out_gemm_kernel(
    const __bf16* __restrict__ A, const __bf16* __restrict__ WT,
    const float* __restrict__ bias, float* __restrict__ Y) {
  constexpr int ALD = 48, BLD = 48;
  constexpr int ASZ = 256 * ALD, BSZ = 64 * BLD;
  __shared__ __align__(16) __bf16 As[2 * ASZ];
  __shared__ __align__(16) __bf16 BsT[2 * BSZ];

  const int tid  = threadIdx.x;
  const int wave = tid >> 5, lane = tid & 31;
  const int col  = lane & 15, halfl = lane >> 4;
  const int nB = blockIdx.x;        // 0..15
  const int mB = blockIdx.y;        // 0..63

  v8f acc[2][4];
#pragma unroll
  for (int tt = 0; tt < 2; ++tt)
#pragma unroll
    for (int tc = 0; tc < 4; ++tc) acc[tt][tc] = vzero8();

  float4 av[4], wv;
#pragma unroll
  for (int it = 0; it < 4; ++it) {
    const int f = tid + it * 256, r = f >> 2, c8 = f & 3;
    av[it] = *(const float4*)(A + (size_t)(mB * 256 + r) * 1024 + c8 * 8);
  }
  { const int n = tid >> 2, c8 = tid & 3;
    wv = *(const float4*)(WT + (size_t)(nB * 64 + n) * 1024 + c8 * 8); }
#pragma unroll
  for (int it = 0; it < 4; ++it) {
    const int f = tid + it * 256, r = f >> 2, c8 = f & 3;
    *(float4*)(As + r * ALD + c8 * 8) = av[it];
  }
  { const int n = tid >> 2, c8 = tid & 3;
    *(float4*)(BsT + n * BLD + c8 * 8) = wv; }
  __syncthreads();

  for (int ks = 0; ks < 32; ++ks) {
    const int cur = ks & 1;
    if (ks + 1 < 32) {
      const int kk = (ks + 1) * 32;
#pragma unroll
      for (int it = 0; it < 4; ++it) {
        const int f = tid + it * 256, r = f >> 2, c8 = f & 3;
        av[it] = *(const float4*)(A + (size_t)(mB * 256 + r) * 1024 + kk + c8 * 8);
      }
      const int n = tid >> 2, c8 = tid & 3;
      wv = *(const float4*)(WT + (size_t)(nB * 64 + n) * 1024 + kk + c8 * 8);
    }
    {
      v16bf a0 = load_frag_a(As + cur * ASZ + (wave * 2 + 0) * 16 * ALD, ALD);
      v16bf a1 = load_frag_a(As + cur * ASZ + (wave * 2 + 1) * 16 * ALD, ALD);
#pragma unroll
      for (int tc = 0; tc < 4; ++tc) {
        v16bf b = load_frag_bT(BsT + cur * BSZ + tc * 16 * BLD, BLD);
        acc[0][tc] = wmma_bf16(a0, b, acc[0][tc]);
        acc[1][tc] = wmma_bf16(a1, b, acc[1][tc]);
      }
    }
    if (ks + 1 < 32) {
      const int nxt = cur ^ 1;
#pragma unroll
      for (int it = 0; it < 4; ++it) {
        const int f = tid + it * 256, r = f >> 2, c8 = f & 3;
        *(float4*)(As + nxt * ASZ + r * ALD + c8 * 8) = av[it];
      }
      const int n = tid >> 2, c8 = tid & 3;
      *(float4*)(BsT + nxt * BSZ + n * BLD + c8 * 8) = wv;
    }
    __syncthreads();
  }

#pragma unroll
  for (int tt = 0; tt < 2; ++tt) {
#pragma unroll
    for (int tc = 0; tc < 4; ++tc) {
      const int cg = nB * 64 + tc * 16 + col;
      const float bv = bias[cg];
#pragma unroll
      for (int i = 0; i < 8; ++i) {
        const int rg = mB * 256 + (wave * 2 + tt) * 16 + halfl * 8 + i;
        Y[(size_t)rg * 1024 + cg] = acc[tt][tc][i] + bv;
      }
    }
  }
}

// ---------------------------------------------------------------------------
extern "C" void kernel_launch(void* const* d_in, const int* in_sizes, int n_in,
                              void* d_out, int out_size, void* d_ws, size_t ws_size,
                              hipStream_t stream) {
  (void)in_sizes; (void)n_in; (void)out_size; (void)ws_size;
  const float* x    = (const float*)d_in[0];   // (4,4096,1024)
  const float* Wqkv = (const float*)d_in[1];   // (1024,3072)
  const float* bqkv = (const float*)d_in[2];   // (3072,)
  const float* Wout = (const float*)d_in[3];   // (1024,1024)
  const float* bout = (const float*)d_in[4];   // (1024,)
  float* y = (float*)d_out;                    // (4,4096,1024) f32

  const size_t NTD = (size_t)16384 * 1024;     // B*L*D elements
  __bf16* q  = (__bf16*)d_ws;                  // [bh][t][e]   32MB
  __bf16* k  = q  + NTD;                       // [bh][t][e]   32MB
  __bf16* kT = k  + NTD;                       // [bh][e][t]   32MB
  __bf16* vT = kT + NTD;                       // [bh][e][t]   32MB
  __bf16* xb = vT + NTD;                       // x as bf16;   32MB
  __bf16* O  = xb;                             // O aliases xb after qkv GEMM
  __bf16* wT = xb + NTD;                       // 6MB region: WqkvT then WoutT

  convert_x_kernel<<<dim3(2048), 256, 0, stream>>>(x, xb);
  convert_wT_kernel<<<dim3(96, 32), 256, 0, stream>>>(Wqkv, wT, 1024, 3072);
  qkv_gemm_kernel<<<dim3(48, 64), 256, 0, stream>>>(xb, wT, bqkv, q, k, kT, vT);
  attn_kernel<<<dim3(64), 256, 0, stream>>>(q, k, kT, vT, O);
  convert_wT_kernel<<<dim3(32, 32), 256, 0, stream>>>(Wout, wT, 1024, 1024);
  out_gemm_kernel<<<dim3(16, 64), 256, 0, stream>>>(O, wT, bout, y);
}
